// Encoder_12618613915965
// MI455X (gfx1250) — matrix operand order repaired and checked by reading, not compiled, not run on hardware
//
#include <hip/hip_runtime.h>

typedef __attribute__((ext_vector_type(16))) __bf16 v16bf;
typedef __attribute__((ext_vector_type(8)))  float  v8f;
typedef __attribute__((address_space(3)))    unsigned short lds_us;

#define BB 4
#define SS 2048
#define DDIM 512
#define HH 8
#define LL 4
#define FFDIM 512
#define DK 64
#define BNEPS 1e-3f

__device__ __forceinline__ unsigned short f2bf(float f) {
  unsigned u = __float_as_uint(f);
  unsigned r = u + 0x7FFFu + ((u >> 16) & 1u);
  return (unsigned short)(r >> 16);
}

__device__ __forceinline__ v8f zero8() {
  v8f z;
  for (int i = 0; i < 8; i++) z[i] = 0.f;
  return z;
}

union Frag16 { uint4 u[2]; v16bf v; };

__device__ __forceinline__ v8f wmma_bf16(v16bf a, v16bf b, v8f c) {
  // D = A(16x32, bf16) * B(32x16, bf16) + C(16x16, f32)
  return __builtin_amdgcn_wmma_f32_16x16x32_bf16(false, a, false, b, (short)0, c,
                                                 false, false);
}

// Async global->LDS copy, 16B per lane, tracked by ASYNCcnt.
__device__ __forceinline__ void async_copy_b128(const unsigned short* gsrc,
                                                unsigned short* ldst) {
  unsigned loff = (unsigned)(unsigned long long)(lds_us*)ldst;  // LDS byte offset
  asm volatile("global_load_async_to_lds_b128 %0, %1, off"
               :: "v"(loff), "v"((unsigned long long)gsrc)
               : "memory");
}
__device__ __forceinline__ void async_wait0() {
  asm volatile("s_wait_asynccnt 0x0" ::: "memory");
}
__device__ __forceinline__ void async_wait2() {
  asm volatile("s_wait_asynccnt 0x2" ::: "memory");
}

// ---------------------------------------------------------------------------
// x = emb[seq] + pos ; fp32 into d_out (residual stream) + bf16 copy
// ---------------------------------------------------------------------------
__global__ void embed_kernel(const int* __restrict__ seq,
                             const float* __restrict__ emb,
                             const float* __restrict__ pos,
                             float* __restrict__ x,
                             unsigned short* __restrict__ xb) {
  int tok = blockIdx.x;
  int s = tok & (SS - 1);
  int id = seq[tok];
  int j = threadIdx.x * 4;
  float4 e = *(const float4*)(emb + (size_t)id * DDIM + j);
  float4 p = *(const float4*)(pos + (size_t)s * DDIM + j);
  float4 v = make_float4(e.x + p.x, e.y + p.y, e.z + p.z, e.w + p.w);
  *(float4*)(x + (size_t)tok * DDIM + j) = v;
  ushort4 h = make_ushort4(f2bf(v.x), f2bf(v.y), f2bf(v.z), f2bf(v.w));
  *(ushort4*)(xb + (size_t)tok * DDIM + j) = h;
}

// ---------------------------------------------------------------------------
// One-time weight transform: fp32 [L][K][N] -> bf16 [L][N][K] (transposed)
// so WMMA B-fragments become contiguous 16B reads. 32x32 LDS-tiled.
// ---------------------------------------------------------------------------
__global__ void cvtT_kernel(const float* __restrict__ in,
                            unsigned short* __restrict__ out, int K, int N) {
  __shared__ float t[32][33];
  int l = blockIdx.z;
  int k0 = blockIdx.x * 32, n0 = blockIdx.y * 32;
  int tx = threadIdx.x & 31, ty = threadIdx.x >> 5;
  const float* src = in + (size_t)l * K * N;
  unsigned short* dst = out + (size_t)l * K * N;
  for (int r = ty; r < 32; r += 8)
    t[r][tx] = src[(size_t)(k0 + r) * N + n0 + tx];
  __syncthreads();
  for (int r = ty; r < 32; r += 8)
    dst[(size_t)(n0 + r) * K + k0 + tx] = f2bf(t[tx][r]);
}

// ---------------------------------------------------------------------------
// GEMM: C[M,N] = A_bf16[M,K] @ Wt_bf16[N,K]^T (+bias), fused epilogue.
// mode 0: bias; 1: bias+ReLU; 2: bias+residual+BatchNorm.
// outBT: optional per-head transposed bf16 store [B][H][dk][S] (for V).
// 8 waves; block 128x128; wave 64x32; K-step 32.
// A tile: double-buffered LDS filled by async-to-LDS DMA (ASYNCcnt).
// B frags: straight from global (weights are L2-resident, pre-transposed).
// ---------------------------------------------------------------------------
__global__ __launch_bounds__(256) void gemm_bf16_kernel(
    const unsigned short* __restrict__ A, const unsigned short* __restrict__ Wt,
    const float* __restrict__ bias, float* __restrict__ outF,
    unsigned short* __restrict__ outB, unsigned short* __restrict__ outBT,
    const float* __restrict__ residual, const float* __restrict__ gamma,
    const float* __restrict__ beta, const float* __restrict__ mean,
    const float* __restrict__ var, int M, int N, int K, int mode) {
  __shared__ unsigned short lA[2][128][40];  // 80B rows, 16B aligned chunks

  const int tid = threadIdx.x;
  const int lane = tid & 31;
  const int w = tid >> 5;
  const int wm = w & 1, wn = w >> 1;
  const int ln16 = lane & 15;
  const int half = lane >> 4;
  const int m0 = blockIdx.x * 128;
  const int n0 = blockIdx.y * 128;

  // per-thread async staging coords (2 x 16B chunks per thread = 8KB tile)
  const int e0 = (tid * 2) * 8;
  const int sr0 = e0 >> 5, sc0 = e0 & 31;
  const int e1 = (tid * 2 + 1) * 8;
  const int sr1 = e1 >> 5, sc1 = e1 & 31;
  const unsigned short* a0 = A + (size_t)(m0 + sr0) * K + sc0;
  const unsigned short* a1 = A + (size_t)(m0 + sr1) * K + sc1;

  // B fragment row pointers (per lane): row = n, contiguous k
  const unsigned short* wr0 = Wt + (size_t)(n0 + wn * 32 + ln16) * K + half * 16;
  const unsigned short* wr1 = wr0 + (size_t)16 * K;

  v8f acc[4][2];
  for (int i = 0; i < 4; i++)
    for (int j = 0; j < 2; j++) acc[i][j] = zero8();

  const int NIT = K >> 5;
  // prologue: stage tile 0
  async_copy_b128(a0, &lA[0][sr0][sc0]);
  async_copy_b128(a1, &lA[0][sr1][sc1]);

  for (int it = 0; it < NIT; it++) {
    const int buf = it & 1;
    const int kk = it << 5;
    if (it + 1 < NIT) {
      async_copy_b128(a0 + kk + 32, &lA[buf ^ 1][sr0][sc0]);
      async_copy_b128(a1 + kk + 32, &lA[buf ^ 1][sr1][sc1]);
      async_wait2();  // retire tile `it` copies; keep next tile in flight
    } else {
      async_wait0();
    }
    __syncthreads();

    Frag16 a[4], b[2];
    for (int mf = 0; mf < 4; mf++) {
      int row = wm * 64 + mf * 16 + ln16;
      a[mf].u[0] = *(const uint4*)(&lA[buf][0][0] + row * 40 + half * 16);
      a[mf].u[1] = *(const uint4*)(&lA[buf][0][0] + row * 40 + half * 16 + 8);
    }
    b[0].u[0] = *(const uint4*)(wr0 + kk);
    b[0].u[1] = *(const uint4*)(wr0 + kk + 8);
    b[1].u[0] = *(const uint4*)(wr1 + kk);
    b[1].u[1] = *(const uint4*)(wr1 + kk + 8);

    for (int mf = 0; mf < 4; mf++)
      for (int nf = 0; nf < 2; nf++)
        acc[mf][nf] = wmma_bf16(a[mf].v, b[nf].v, acc[mf][nf]);
    __syncthreads();  // everyone done reading buf before it is refilled
  }

  // Epilogue. C layout: row = r + 8*half, col = lane%16.
  for (int mf = 0; mf < 4; mf++) {
    for (int nf = 0; nf < 2; nf++) {
      int gn = n0 + wn * 32 + nf * 16 + ln16;
      float bs = bias[gn];
      float g = 0.f, bt = 0.f, mu = 0.f, rstd = 0.f;
      if (mode == 2) {
        g = gamma[gn]; bt = beta[gn]; mu = mean[gn];
        rstd = rsqrtf(var[gn] + BNEPS);
      }
      int gmb = m0 + wm * 64 + mf * 16 + half * 8;
      float vals[8];
      for (int r = 0; r < 8; r++) {
        int gm = gmb + r;
        float v = acc[mf][nf][r] + bs;
        if (mode == 1) {
          v = v > 0.f ? v : 0.f;
        } else if (mode == 2) {
          v = v + residual[(size_t)gm * N + gn];
          v = (v - mu) * rstd * g + bt;
        }
        vals[r] = v;
      }
      if (outF)
        for (int r = 0; r < 8; r++) outF[(size_t)(gmb + r) * N + gn] = vals[r];
      if (outB)
        for (int r = 0; r < 8; r++)
          outB[(size_t)(gmb + r) * N + gn] = f2bf(vals[r]);
      if (outBT) {
        // [B][H][dk][S]: 8 consecutive s positions -> two 8B stores
        int bi = gmb >> 11, si = gmb & (SS - 1);
        int hi = gn >> 6, dk = gn & (DK - 1);
        ushort4 p0 = make_ushort4(f2bf(vals[0]), f2bf(vals[1]), f2bf(vals[2]),
                                  f2bf(vals[3]));
        ushort4 p1 = make_ushort4(f2bf(vals[4]), f2bf(vals[5]), f2bf(vals[6]),
                                  f2bf(vals[7]));
        unsigned short* d =
            outBT + (((size_t)bi * HH + hi) * DK + dk) * SS + si;
        *(uint2*)(d) = *(uint2*)&p0;
        *(uint2*)(d + 4) = *(uint2*)&p1;
      }
    }
  }
}

// ---------------------------------------------------------------------------
// Flash attention, LDS-free operand streaming:
// grid (S/64, H, B), 128 threads = 4 independent waves x 16 query rows.
// K frags read from [B,S,D] rows; V frags from per-head transposed [B,H,dk,S].
// Only the P (C-layout -> A-layout) bounce uses LDS, fenced by s_wait_dscnt.
// ---------------------------------------------------------------------------
__global__ __launch_bounds__(128) void flash_attn_kernel(
    const unsigned short* __restrict__ Q, const unsigned short* __restrict__ Kp,
    const unsigned short* __restrict__ Vt, const float* __restrict__ pm,
    unsigned short* __restrict__ ctx) {
  __shared__ unsigned short lP[4][16][40];  // per-wave P tile [m][key]

  const int tid = threadIdx.x;
  const int lane = tid & 31;
  const int w = tid >> 5;
  const int ln16 = lane & 15;
  const int half = lane >> 4;
  const int b = blockIdx.z, h = blockIdx.y;
  const int qbase = blockIdx.x * 64 + w * 16;

  const unsigned short* Qh = Q + ((size_t)b * SS) * DDIM + h * DK;
  const unsigned short* Kh = Kp + ((size_t)b * SS) * DDIM + h * DK;

  // Preload Q A-fragments for both dk halves (reused for all key blocks)
  Frag16 qa[2];
  {
    const unsigned short* qr = Qh + (size_t)(qbase + ln16) * DDIM;
    for (int t = 0; t < 2; t++) {
      qa[t].u[0] = *(const uint4*)(qr + t * 32 + half * 16);
      qa[t].u[1] = *(const uint4*)(qr + t * 32 + half * 16 + 8);
    }
  }
  // K fragment row pointers: lane -> key row (lo / hi 16-key tile)
  const unsigned short* krl = Kh + (size_t)ln16 * DDIM + half * 16;
  const unsigned short* krh = krl + (size_t)16 * DDIM;
  // V fragment row pointers: lane -> dk row of transposed V
  const unsigned short* vr[4];
  for (int nf = 0; nf < 4; nf++)
    vr[nf] = Vt + (((size_t)b * HH + h) * DK + nf * 16 + ln16) * SS + half * 16;

  float mrow[8], lrow[8];
  v8f o[4];
  for (int r = 0; r < 8; r++) { mrow[r] = -1e30f; lrow[r] = 0.f; }
  for (int i = 0; i < 4; i++) o[i] = zero8();

  const float scale = 0.125f;  // 1/sqrt(dk=64)

  for (int kb = 0; kb < SS; kb += 32) {
    // Scores: S(16q x 32keys), two 16-key C frags, dk in two 32-chunks
    v8f s0 = zero8(), s1 = zero8();
    for (int t = 0; t < 2; t++) {
      Frag16 bk;
      size_t koff = (size_t)kb * DDIM + t * 32;
      bk.u[0] = *(const uint4*)(krl + koff);
      bk.u[1] = *(const uint4*)(krl + koff + 8);
      s0 = wmma_bf16(qa[t].v, bk.v, s0);
      bk.u[0] = *(const uint4*)(krh + koff);
      bk.u[1] = *(const uint4*)(krh + koff + 8);
      s1 = wmma_bf16(qa[t].v, bk.v, s1);
    }

    // Mask + online softmax (reference: s*scale*m, then ==0 -> -1e9)
    float cm0 = pm[(size_t)b * SS + kb + ln16];
    float cm1 = pm[(size_t)b * SS + kb + 16 + ln16];
    for (int r = 0; r < 8; r++) {
      float a0 = s0[r] * scale * cm0; if (a0 == 0.f) a0 = -1e9f;
      float a1 = s1[r] * scale * cm1; if (a1 == 0.f) a1 = -1e9f;
      float loc = fmaxf(a0, a1);
      for (int dd = 1; dd < 16; dd <<= 1)
        loc = fmaxf(loc, __shfl_xor(loc, dd, 32));
      float mn = fmaxf(mrow[r], loc);
      float corr = __expf(mrow[r] - mn);
      mrow[r] = mn;
      float p0 = __expf(a0 - mn);
      float p1 = __expf(a1 - mn);
      float rs = p0 + p1;
      for (int dd = 1; dd < 16; dd <<= 1) rs += __shfl_xor(rs, dd, 32);
      lrow[r] = lrow[r] * corr + rs;
      for (int nf = 0; nf < 4; nf++) o[nf][r] *= corr;
      int m = r + half * 8;
      lP[w][m][ln16] = f2bf(p0);
      lP[w][m][16 + ln16] = f2bf(p1);
    }
    // wave-private LDS bounce: C-layout -> A-layout
    asm volatile("s_wait_dscnt 0x0" ::: "memory");
    Frag16 pa;
    pa.u[0] = *(const uint4*)(&lP[w][0][0] + ln16 * 40 + half * 16);
    pa.u[1] = *(const uint4*)(&lP[w][0][0] + ln16 * 40 + half * 16 + 8);
    for (int nf = 0; nf < 4; nf++) {
      Frag16 bv;
      bv.u[0] = *(const uint4*)(vr[nf] + kb);
      bv.u[1] = *(const uint4*)(vr[nf] + kb + 8);
      o[nf] = wmma_bf16(pa.v, bv.v, o[nf]);
    }
  }

  // Normalize and store ctx (bf16, merged-head layout [B,S,D])
  for (int nf = 0; nf < 4; nf++) {
    for (int r = 0; r < 8; r++) {
      int qrow = qbase + r + half * 8;
      float v = o[nf][r] / lrow[r];
      ctx[((size_t)b * SS + qrow) * DDIM + h * DK + nf * 16 + ln16] = f2bf(v);
    }
  }
}

// ---------------------------------------------------------------------------
extern "C" void kernel_launch(void* const* d_in, const int* in_sizes, int n_in,
                              void* d_out, int out_size, void* d_ws,
                              size_t ws_size, hipStream_t stream) {
  const int* seq = (const int*)d_in[0];
  const float* pm = (const float*)d_in[1];
  const float* pos = (const float*)d_in[2];
  const float* emb = (const float*)d_in[3];
  const float* wq = (const float*)d_in[4];
  const float* bq = (const float*)d_in[5];
  const float* wk = (const float*)d_in[6];
  const float* bk = (const float*)d_in[7];
  const float* wv = (const float*)d_in[8];
  const float* bv = (const float*)d_in[9];
  const float* wo = (const float*)d_in[10];
  const float* bo = (const float*)d_in[11];
  const float* an_g = (const float*)d_in[12];
  const float* an_b = (const float*)d_in[13];
  const float* an_m = (const float*)d_in[14];
  const float* an_v = (const float*)d_in[15];
  const float* d1w = (const float*)d_in[16];
  const float* d1b = (const float*)d_in[17];
  const float* d2w = (const float*)d_in[18];
  const float* d2b = (const float*)d_in[19];
  const float* fn_g = (const float*)d_in[20];
  const float* fn_b = (const float*)d_in[21];
  const float* fn_m = (const float*)d_in[22];
  const float* fn_v = (const float*)d_in[23];

  float* x = (float*)d_out;  // fp32 residual stream lives in d_out

  unsigned short* ws16 = (unsigned short*)d_ws;
  const size_t AB = (size_t)BB * SS * DDIM;  // 4 Mi elements
  const size_t WSZ = (size_t)LL * DDIM * DDIM;
  unsigned short* x_b = ws16;
  unsigned short* q_b = x_b + AB;
  unsigned short* k_b = q_b + AB;
  unsigned short* v_t = k_b + AB;  // per-head transposed V [B][H][dk][S]
  unsigned short* c_b = v_t + AB;
  unsigned short* s_b = c_b + AB;
  unsigned short* f_b = s_b + AB;
  unsigned short* wq_t = f_b + AB;  // all weights bf16, transposed [N][K]
  unsigned short* wk_t = wq_t + WSZ;
  unsigned short* wv_t = wk_t + WSZ;
  unsigned short* wo_t = wv_t + WSZ;
  unsigned short* d1_t = wo_t + WSZ;
  unsigned short* d2_t = d1_t + (size_t)LL * DDIM * FFDIM;
  float* sub_f = (float*)(d2_t + (size_t)LL * FFDIM * DDIM);

  dim3 tgrid(DDIM / 32, DDIM / 32, LL);
  cvtT_kernel<<<tgrid, 256, 0, stream>>>(wq, wq_t, DDIM, DDIM);
  cvtT_kernel<<<tgrid, 256, 0, stream>>>(wk, wk_t, DDIM, DDIM);
  cvtT_kernel<<<tgrid, 256, 0, stream>>>(wv, wv_t, DDIM, DDIM);
  cvtT_kernel<<<tgrid, 256, 0, stream>>>(wo, wo_t, DDIM, DDIM);
  cvtT_kernel<<<tgrid, 256, 0, stream>>>(d1w, d1_t, DDIM, FFDIM);
  cvtT_kernel<<<tgrid, 256, 0, stream>>>(d2w, d2_t, FFDIM, DDIM);

  embed_kernel<<<BB * SS, 128, 0, stream>>>(seq, emb, pos, x, x_b);

  dim3 ggrid((BB * SS) / 128, DDIM / 128);
  auto gemm = [&](const unsigned short* A, const unsigned short* Wt,
                  const float* bias, float* oF, unsigned short* oB,
                  unsigned short* oBT, const float* res, const float* g,
                  const float* bt, const float* mu, const float* vr, int mode) {
    gemm_bf16_kernel<<<ggrid, 256, 0, stream>>>(A, Wt, bias, oF, oB, oBT, res,
                                                g, bt, mu, vr, BB * SS, DDIM,
                                                DDIM, mode);
  };

  for (int i = 0; i < LL; i++) {
    const size_t wOff = (size_t)i * DDIM * DDIM;
    // Q/K projections (bf16), V projection stored per-head transposed
    gemm(x_b, wq_t + wOff, bq + i * DDIM, nullptr, q_b, nullptr, nullptr,
         nullptr, nullptr, nullptr, nullptr, 0);
    gemm(x_b, wk_t + wOff, bk + i * DDIM, nullptr, k_b, nullptr, nullptr,
         nullptr, nullptr, nullptr, nullptr, 0);
    gemm(x_b, wv_t + wOff, bv + i * DDIM, nullptr, nullptr, v_t, nullptr,
         nullptr, nullptr, nullptr, nullptr, 0);
    // Attention
    flash_attn_kernel<<<dim3(SS / 64, HH, BB), 128, 0, stream>>>(q_b, k_b, v_t,
                                                                 pm, c_b);
    // Output projection + residual + BN -> sub (fp32 + bf16)
    gemm(c_b, wo_t + wOff, bo + i * DDIM, sub_f, s_b, nullptr, x,
         an_g + i * DDIM, an_b + i * DDIM, an_m + i * DDIM, an_v + i * DDIM, 2);
    // FFN up + ReLU (bf16 only)
    gemm(s_b, d1_t + wOff, d1b + i * FFDIM, nullptr, f_b, nullptr, nullptr,
         nullptr, nullptr, nullptr, nullptr, 1);
    // FFN down + residual + BN -> new x (fp32 into d_out + bf16)
    gemm(f_b, d2_t + wOff, d2b + i * DDIM, x, x_b, nullptr, sub_f,
         fn_g + i * DDIM, fn_b + i * DDIM, fn_m + i * DDIM, fn_v + i * DDIM, 2);
  }
}
